// TransformerBlock_78932908966067
// MI455X (gfx1250) — compile-verified, matrix operand order checked by reading, not run on hardware
//
#include <hip/hip_runtime.h>
#include <hip/hip_bf16.h>

// ---------------------------------------------------------------------------
// Transformer block for MI455X (gfx1250, wave32, WMMA 16x16x32 f16->f32).
// f16 compute / f32 accumulate; all GEMMs via v_wmma_f32_16x16x32_f16.
// GEMM tiles staged with GLOBAL_LOAD_ASYNC_TO_LDS + ASYNCcnt double buffering
// (builtins confirmed present on this toolchain; guarded fallback retained).
// ---------------------------------------------------------------------------

#define B_  4
#define S_  2048
#define D_  1024
#define H_  16
#define HD_ 64
#define FF_ 4096
#define M_  (B_ * S_)      // 8192 rows

#if defined(__has_builtin)
#if __has_builtin(__builtin_amdgcn_global_load_async_to_lds_b128) && \
    __has_builtin(__builtin_amdgcn_s_wait_asynccnt)
#define CDNA5_ASYNC 1
#endif
#endif
#ifndef CDNA5_ASYNC
#define CDNA5_ASYNC 0
#endif

typedef _Float16 h16 __attribute__((ext_vector_type(16)));
typedef _Float16 h8  __attribute__((ext_vector_type(8)));
typedef float    f8  __attribute__((ext_vector_type(8)));
typedef int      i4v __attribute__((ext_vector_type(4)));

__device__ __forceinline__ f8 wmma_f16(h16 a, h16 b, f8 c) {
  // D = A(16x32 f16) * B(32x16 f16) + C(16x16 f32)
  return __builtin_amdgcn_wmma_f32_16x16x32_f16(
      /*neg_a=*/false, a, /*neg_b=*/false, b,
      /*c_mod=*/(short)0, c, /*reuse_a=*/false, /*reuse_b=*/false);
}

// Build a 16-half A/B fragment from two contiguous 8-half (16B) chunks.
__device__ __forceinline__ h16 mkfrag(const _Float16* p0, const _Float16* p1) {
  h8 a = *(const h8*)p0;
  h8 b = *(const h8*)p1;
  h16 r;
#pragma unroll
  for (int i = 0; i < 8; ++i) { r[i] = a[i]; r[8 + i] = b[i]; }
  return r;
}

__device__ __forceinline__ f8 fzero8() {
  f8 z = {0.f, 0.f, 0.f, 0.f, 0.f, 0.f, 0.f, 0.f};
  return z;
}

#if CDNA5_ASYNC
// 16B async DMA: global -> LDS, tracked with ASYNCcnt (no VGPR round trip).
__device__ __forceinline__ void async_cp16(const _Float16* g, _Float16* l) {
  __builtin_amdgcn_global_load_async_to_lds_b128(
      (__attribute__((address_space(1))) i4v*)g,
      (__attribute__((address_space(3))) i4v*)l, 0, 0);
}
#endif

// ---------------------------------------------------------------------------
// Elementwise converts / weight transposes (one-time prep, bandwidth-trivial)
// ---------------------------------------------------------------------------
__global__ void __launch_bounds__(256) cvt_f32_f16_kernel(const float* __restrict__ src,
                                                          _Float16* __restrict__ dst,
                                                          size_t n) {
  size_t i = (size_t)blockIdx.x * blockDim.x + threadIdx.x;
  if (i < n) dst[i] = (_Float16)src[i];
}

// Fused QKV weight: Wt[n][k] (n in [0,3D)) with n<D -> wq[h][k][f], etc.
__global__ void __launch_bounds__(256) qkvw_kernel(const float* __restrict__ wq,
                                                   const float* __restrict__ wk,
                                                   const float* __restrict__ wv,
                                                   _Float16* __restrict__ Wt) {
  size_t i = (size_t)blockIdx.x * blockDim.x + threadIdx.x;
  size_t total = (size_t)3 * D_ * D_;
  if (i >= total) return;
  size_t n = i / D_;
  size_t k = i % D_;
  int slab = (int)(n >> 10);
  int np   = (int)(n & 1023);
  int hh   = np >> 6;
  int f    = np & 63;
  const float* w = (slab == 0) ? wq : (slab == 1) ? wk : wv;
  Wt[i] = (_Float16)w[((size_t)hh * D_ + k) * HD_ + f];
}

// dst[n][k] = src[k][n], src is [K][N] f32 row-major, dst is [N][K] f16.
__global__ void __launch_bounds__(256) transpose_kernel(const float* __restrict__ src,
                                                        _Float16* __restrict__ dst,
                                                        int K, int N) {
  size_t i = (size_t)blockIdx.x * blockDim.x + threadIdx.x;
  size_t total = (size_t)K * N;
  if (i >= total) return;
  size_t n = i / K;
  size_t k = i % K;
  dst[i] = (_Float16)src[k * (size_t)N + n];
}

// ---------------------------------------------------------------------------
// WMMA GEMM tiling constants: 128x128 tile, BK=32, 256 threads = 8 waves as
// 4x2; each wave computes a 32x64 subtile = 2x4 WMMA accumulators.
// ---------------------------------------------------------------------------
#define BM 128
#define BN 128
#define BK 32
#define LSTR 40   // LDS row stride in halfs (80B) -> bank-conflict padding

// Shared GEMM mainloop body (macro so both kernels get identical codegen).
// Defines/updates: acc[2][4] from As/Bs tiles.
#define GEMM_FRAG_AND_MMA(ASBUF, BSBUF)                                        \
  do {                                                                         \
    h16 af[2], bf[4];                                                          \
    _Pragma("unroll") for (int tm = 0; tm < 2; ++tm) {                         \
      int row = waveM * 32 + tm * 16 + lr;                                     \
      af[tm] = mkfrag(&(ASBUF)[row * LSTR + base0],                            \
                      &(ASBUF)[row * LSTR + base0 + 16]);                      \
    }                                                                          \
    _Pragma("unroll") for (int tn = 0; tn < 4; ++tn) {                         \
      int nn = waveN * 64 + tn * 16 + lr;                                      \
      bf[tn] = mkfrag(&(BSBUF)[nn * LSTR + base0],                             \
                      &(BSBUF)[nn * LSTR + base0 + 16]);                       \
    }                                                                          \
    _Pragma("unroll") for (int tm = 0; tm < 2; ++tm)                           \
        _Pragma("unroll") for (int tn = 0; tn < 4; ++tn)                       \
            acc[tm][tn] = wmma_f16(af[tm], bf[tn], acc[tm][tn]);               \
  } while (0)

// ---------------------------------------------------------------------------
// Generic WMMA GEMM: C[M,N] = act(A[M,K](f16) * Bt[N,K](f16)^T + bias)
// ---------------------------------------------------------------------------
__global__ void __launch_bounds__(256) gemm_wmma_kernel(
    const _Float16* __restrict__ A, const _Float16* __restrict__ Bt,
    const float* __restrict__ bias, float* __restrict__ outF32,
    _Float16* __restrict__ outF16, int M, int N, int K, int act_gelu) {
  const int tid  = threadIdx.x;
  const int lane = tid & 31;
  const int wid  = tid >> 5;
  const int lr   = lane & 15;
  const int hi   = lane >> 4;
  const int base0 = hi * 8;
  const int waveM = wid >> 1;   // 0..3
  const int waveN = wid & 1;    // 0..1

  const size_t m0 = (size_t)blockIdx.y * BM;
  const size_t n0 = (size_t)blockIdx.x * BN;

  const int lrow = tid >> 1;          // 0..127
  const int lseg = (tid & 1) * 16;    // 0 / 16 halfs within BK=32

  const _Float16* Arow = A  + (m0 + lrow) * (size_t)K + lseg;
  const _Float16* Brow = Bt + (n0 + lrow) * (size_t)K + lseg;

  f8 acc[2][4];
#pragma unroll
  for (int tm = 0; tm < 2; ++tm)
#pragma unroll
    for (int tn = 0; tn < 4; ++tn) acc[tm][tn] = fzero8();

#if CDNA5_ASYNC
  // Double-buffered async global->LDS pipeline on ASYNCcnt.
  __shared__ __align__(16) _Float16 As[2][BM * LSTR];
  __shared__ __align__(16) _Float16 Bs[2][BN * LSTR];
  {
    // prologue: tile 0 into buffer 0 (4 async b128 per thread)
    async_cp16(Arow + 0, &As[0][lrow * LSTR + lseg]);
    async_cp16(Arow + 8, &As[0][lrow * LSTR + lseg + 8]);
    async_cp16(Brow + 0, &Bs[0][lrow * LSTR + lseg]);
    async_cp16(Brow + 8, &Bs[0][lrow * LSTR + lseg + 8]);
  }
  int it = 0;
  for (int k0 = 0; k0 < K; k0 += BK, ++it) {
    const int cur = it & 1;
    const bool more = (k0 + BK) < K;
    if (more) {  // overlap next tile's DMA with this tile's WMMAs
      const int nxt = cur ^ 1;
      async_cp16(Arow + k0 + BK,     &As[nxt][lrow * LSTR + lseg]);
      async_cp16(Arow + k0 + BK + 8, &As[nxt][lrow * LSTR + lseg + 8]);
      async_cp16(Brow + k0 + BK,     &Bs[nxt][lrow * LSTR + lseg]);
      async_cp16(Brow + k0 + BK + 8, &Bs[nxt][lrow * LSTR + lseg + 8]);
      // Retire this wave's 4 copies for the current buffer (in-order; 4 newer
      // ones may remain outstanding), then make them visible workgroup-wide.
      __builtin_amdgcn_s_wait_asynccnt(4);
    } else {
      __builtin_amdgcn_s_wait_asynccnt(0);
    }
    __syncthreads();
    GEMM_FRAG_AND_MMA(As[cur], Bs[cur]);
    __syncthreads();
  }
#else
  // Fallback: synchronous staging through VGPRs, single buffer.
  __shared__ __align__(16) _Float16 As[1][BM * LSTR];
  __shared__ __align__(16) _Float16 Bs[1][BN * LSTR];
  for (int k0 = 0; k0 < K; k0 += BK) {
    h8 a0 = *(const h8*)(Arow + k0);
    h8 a1 = *(const h8*)(Arow + k0 + 8);
    h8 b0 = *(const h8*)(Brow + k0);
    h8 b1 = *(const h8*)(Brow + k0 + 8);
    *(h8*)(&As[0][lrow * LSTR + lseg])     = a0;
    *(h8*)(&As[0][lrow * LSTR + lseg + 8]) = a1;
    *(h8*)(&Bs[0][lrow * LSTR + lseg])     = b0;
    *(h8*)(&Bs[0][lrow * LSTR + lseg + 8]) = b1;
    if (k0 + BK < K) {
      __builtin_prefetch(Arow + k0 + BK, 0, 1);
      __builtin_prefetch(Brow + k0 + BK, 0, 1);
    }
    __syncthreads();
    GEMM_FRAG_AND_MMA(As[0], Bs[0]);
    __syncthreads();
  }
#endif

#pragma unroll
  for (int tm = 0; tm < 2; ++tm)
#pragma unroll
    for (int tn = 0; tn < 4; ++tn) {
      size_t col = n0 + waveN * 64 + tn * 16 + lr;
      float bv = bias ? bias[col] : 0.f;
#pragma unroll
      for (int r = 0; r < 8; ++r) {
        size_t row = m0 + waveM * 32 + tm * 16 + r + 8 * hi;
        float v = acc[tm][tn][r] + bv;
        if (act_gelu) v = 0.5f * v * (1.f + erff(v * 0.70710678118f));
        if (outF32) outF32[row * (size_t)N + col] = v;
        if (outF16) outF16[row * (size_t)N + col] = (_Float16)v;
      }
    }
}

// ---------------------------------------------------------------------------
// QKV GEMM: same tiling, special epilogue storing Q,K as [B,H,S,HD] and
// V transposed as [B,H,HD,S] (so attention's PV B-fragments are contiguous).
// ---------------------------------------------------------------------------
__global__ void __launch_bounds__(256) qkv_gemm_kernel(
    const _Float16* __restrict__ A, const _Float16* __restrict__ Wt,
    const float* __restrict__ bq, const float* __restrict__ bk,
    const float* __restrict__ bv, _Float16* __restrict__ Q,
    _Float16* __restrict__ Kh, _Float16* __restrict__ Vt) {
  const int tid  = threadIdx.x;
  const int lane = tid & 31;
  const int wid  = tid >> 5;
  const int lr   = lane & 15;
  const int hi   = lane >> 4;
  const int base0 = hi * 8;
  const int waveM = wid >> 1;
  const int waveN = wid & 1;
  const int K = D_;

  const size_t m0 = (size_t)blockIdx.y * BM;
  const size_t n0 = (size_t)blockIdx.x * BN;

  const int lrow = tid >> 1;
  const int lseg = (tid & 1) * 16;

  const _Float16* Arow = A  + (m0 + lrow) * (size_t)K + lseg;
  const _Float16* Brow = Wt + (n0 + lrow) * (size_t)K + lseg;

  f8 acc[2][4];
#pragma unroll
  for (int tm = 0; tm < 2; ++tm)
#pragma unroll
    for (int tn = 0; tn < 4; ++tn) acc[tm][tn] = fzero8();

#if CDNA5_ASYNC
  __shared__ __align__(16) _Float16 As[2][BM * LSTR];
  __shared__ __align__(16) _Float16 Bs[2][BN * LSTR];
  {
    async_cp16(Arow + 0, &As[0][lrow * LSTR + lseg]);
    async_cp16(Arow + 8, &As[0][lrow * LSTR + lseg + 8]);
    async_cp16(Brow + 0, &Bs[0][lrow * LSTR + lseg]);
    async_cp16(Brow + 8, &Bs[0][lrow * LSTR + lseg + 8]);
  }
  int it = 0;
  for (int k0 = 0; k0 < K; k0 += BK, ++it) {
    const int cur = it & 1;
    const bool more = (k0 + BK) < K;
    if (more) {
      const int nxt = cur ^ 1;
      async_cp16(Arow + k0 + BK,     &As[nxt][lrow * LSTR + lseg]);
      async_cp16(Arow + k0 + BK + 8, &As[nxt][lrow * LSTR + lseg + 8]);
      async_cp16(Brow + k0 + BK,     &Bs[nxt][lrow * LSTR + lseg]);
      async_cp16(Brow + k0 + BK + 8, &Bs[nxt][lrow * LSTR + lseg + 8]);
      __builtin_amdgcn_s_wait_asynccnt(4);
    } else {
      __builtin_amdgcn_s_wait_asynccnt(0);
    }
    __syncthreads();
    GEMM_FRAG_AND_MMA(As[cur], Bs[cur]);
    __syncthreads();
  }
#else
  __shared__ __align__(16) _Float16 As[1][BM * LSTR];
  __shared__ __align__(16) _Float16 Bs[1][BN * LSTR];
  for (int k0 = 0; k0 < K; k0 += BK) {
    h8 a0 = *(const h8*)(Arow + k0);
    h8 a1 = *(const h8*)(Arow + k0 + 8);
    h8 b0 = *(const h8*)(Brow + k0);
    h8 b1 = *(const h8*)(Brow + k0 + 8);
    *(h8*)(&As[0][lrow * LSTR + lseg])     = a0;
    *(h8*)(&As[0][lrow * LSTR + lseg + 8]) = a1;
    *(h8*)(&Bs[0][lrow * LSTR + lseg])     = b0;
    *(h8*)(&Bs[0][lrow * LSTR + lseg + 8]) = b1;
    if (k0 + BK < K) {
      __builtin_prefetch(Arow + k0 + BK, 0, 1);
      __builtin_prefetch(Brow + k0 + BK, 0, 1);
    }
    __syncthreads();
    GEMM_FRAG_AND_MMA(As[0], Bs[0]);
    __syncthreads();
  }
#endif

#pragma unroll
  for (int tm = 0; tm < 2; ++tm)
#pragma unroll
    for (int tn = 0; tn < 4; ++tn) {
      int col  = (int)n0 + waveN * 64 + tn * 16 + lr;  // 0..3071
      int slab = col >> 10;
      int np   = col & 1023;
      int hh   = np >> 6;
      int f    = np & 63;
      float bvv = (slab == 0) ? bq[np] : (slab == 1) ? bk[np] : bv[np];
#pragma unroll
      for (int r = 0; r < 8; ++r) {
        size_t row = m0 + waveM * 32 + tm * 16 + r + 8 * hi;  // global token
        int b = (int)(row >> 11);      // / S_
        int s = (int)(row & 2047);     // % S_
        _Float16 hv = (_Float16)(acc[tm][tn][r] + bvv);
        if (slab == 0)
          Q[(((size_t)b * H_ + hh) * S_ + s) * HD_ + f] = hv;
        else if (slab == 1)
          Kh[(((size_t)b * H_ + hh) * S_ + s) * HD_ + f] = hv;
        else
          Vt[(((size_t)b * H_ + hh) * HD_ + f) * S_ + s] = hv;
      }
    }
}

// ---------------------------------------------------------------------------
// Causal flash attention. Block = 128 q rows (8 waves x 16 rows), steps of 32
// keys. QK^T: 2 WMMAs; online softmax with shfl_xor row reductions; P goes
// through per-wave LDS to re-layout C->A fragment; PV: 4 WMMAs (HD=64).
// ---------------------------------------------------------------------------
#define PSTR 40

__global__ void __launch_bounds__(256) attn_kernel(
    const _Float16* __restrict__ Q, const _Float16* __restrict__ Kh,
    const _Float16* __restrict__ Vt, _Float16* __restrict__ O) {
  __shared__ __align__(16) _Float16 Plds[8][16 * PSTR];

  const int b = blockIdx.z;
  const int h = blockIdx.y;
  const int tid  = threadIdx.x;
  const int lane = tid & 31;
  const int wid  = tid >> 5;
  const int lr   = lane & 15;
  const int hi   = lane >> 4;
  const int base0 = hi * 8;
  const int qr0  = blockIdx.x * 128 + wid * 16;

  const size_t headQK = ((size_t)(b * H_ + h)) * S_ * HD_;
  const _Float16* Qp = Q  + headQK;
  const _Float16* Kp = Kh + headQK;
  const _Float16* Vp = Vt + ((size_t)(b * H_ + h)) * HD_ * S_;

  h16 qf[2];
#pragma unroll
  for (int c = 0; c < 2; ++c) {
    const _Float16* p = Qp + (size_t)(qr0 + lr) * HD_ + c * 32;
    qf[c] = mkfrag(p + base0, p + base0 + 16);
  }

  f8 acc[4];
#pragma unroll
  for (int tn = 0; tn < 4; ++tn) acc[tn] = fzero8();
  float mi[8], li[8];
#pragma unroll
  for (int r = 0; r < 8; ++r) { mi[r] = -1e30f; li[r] = 0.f; }

  const int kend = blockIdx.x * 128 + 128;  // uniform across waves -> barriers ok
  for (int k0 = 0; k0 < kend; k0 += 32) {
    float sv[2][8];
#pragma unroll
    for (int sub = 0; sub < 2; ++sub) {
      int kb = k0 + sub * 16;
      const _Float16* kp = Kp + (size_t)(kb + lr) * HD_;
      h16 kf0 = mkfrag(kp + base0, kp + base0 + 16);
      h16 kf1 = mkfrag(kp + 32 + base0, kp + 32 + base0 + 16);
      f8 sacc = fzero8();
      sacc = wmma_f16(qf[0], kf0, sacc);
      sacc = wmma_f16(qf[1], kf1, sacc);
      int col = kb + lr;
#pragma unroll
      for (int r = 0; r < 8; ++r) {
        int row = qr0 + r + 8 * hi;
        float v = sacc[r] * 0.125f;  // 1/sqrt(HD)
        sv[sub][r] = (col <= row) ? v : -1e30f;
      }
    }

    float alpha[8], p0a[8], p1a[8];
#pragma unroll
    for (int r = 0; r < 8; ++r) {
      float mx = fmaxf(sv[0][r], sv[1][r]);
      mx = fmaxf(mx, __shfl_xor(mx, 1));
      mx = fmaxf(mx, __shfl_xor(mx, 2));
      mx = fmaxf(mx, __shfl_xor(mx, 4));
      mx = fmaxf(mx, __shfl_xor(mx, 8));
      float mn = fmaxf(mi[r], mx);
      float p0 = __expf(sv[0][r] - mn);
      float p1 = __expf(sv[1][r] - mn);
      float rs = p0 + p1;
      rs += __shfl_xor(rs, 1);
      rs += __shfl_xor(rs, 2);
      rs += __shfl_xor(rs, 4);
      rs += __shfl_xor(rs, 8);
      alpha[r] = __expf(mi[r] - mn);
      li[r] = li[r] * alpha[r] + rs;
      mi[r] = mn;
      p0a[r] = p0;
      p1a[r] = p1;
    }
#pragma unroll
    for (int tn = 0; tn < 4; ++tn)
#pragma unroll
      for (int r = 0; r < 8; ++r) acc[tn][r] *= alpha[r];

    // P (C-layout, f32) -> LDS (row-major f16) so it can be an A fragment.
    _Float16* pw = &Plds[wid][0];
#pragma unroll
    for (int r = 0; r < 8; ++r) {
      pw[(r + 8 * hi) * PSTR + lr]      = (_Float16)p0a[r];
      pw[(r + 8 * hi) * PSTR + 16 + lr] = (_Float16)p1a[r];
    }
    __syncthreads();
    h16 pf = mkfrag(&Plds[wid][lr * PSTR + base0],
                    &Plds[wid][lr * PSTR + base0 + 16]);
#pragma unroll
    for (int tn = 0; tn < 4; ++tn) {
      const _Float16* vp = Vp + (size_t)(tn * 16 + lr) * S_ + k0;
      h16 vf = mkfrag(vp + base0, vp + base0 + 16);
      acc[tn] = wmma_f16(pf, vf, acc[tn]);
    }
    __syncthreads();
  }

#pragma unroll
  for (int tn = 0; tn < 4; ++tn) {
    int f = tn * 16 + lr;
#pragma unroll
    for (int r = 0; r < 8; ++r) {
      int row = qr0 + r + 8 * hi;
      float v = acc[tn][r] / li[r];
      O[((size_t)b * S_ + row) * D_ + h * HD_ + f] = (_Float16)v;
    }
  }
}

// ---------------------------------------------------------------------------
// Residual add + LayerNorm, one row (D=1024) per 256-thread block.
// ---------------------------------------------------------------------------
__global__ void __launch_bounds__(256) add_ln_kernel(
    const float* __restrict__ a, const float* __restrict__ c,
    const float* __restrict__ g, const float* __restrict__ beta,
    float* __restrict__ o32, _Float16* __restrict__ o16) {
  __shared__ float red[256];
  const int t = threadIdx.x;
  const size_t off = (size_t)blockIdx.x * D_;

  float v[4];
  float s = 0.f;
#pragma unroll
  for (int i = 0; i < 4; ++i) {
    int col = t + i * 256;
    float z = a[off + col] + c[off + col];
    v[i] = z;
    s += z;
  }
  red[t] = s;
  __syncthreads();
  for (int w = 128; w > 0; w >>= 1) {
    if (t < w) red[t] += red[t + w];
    __syncthreads();
  }
  float mean = red[0] * (1.f / D_);
  __syncthreads();

  float q = 0.f;
#pragma unroll
  for (int i = 0; i < 4; ++i) {
    float d = v[i] - mean;
    q += d * d;
  }
  red[t] = q;
  __syncthreads();
  for (int w = 128; w > 0; w >>= 1) {
    if (t < w) red[t] += red[t + w];
    __syncthreads();
  }
  float rstd = rsqrtf(red[0] * (1.f / D_) + 1e-5f);

#pragma unroll
  for (int i = 0; i < 4; ++i) {
    int col = t + i * 256;
    float y = (v[i] - mean) * rstd * g[col] + beta[col];
    o32[off + col] = y;
    if (o16) o16[off + col] = (_Float16)y;
  }
}

// ---------------------------------------------------------------------------
// Host-side orchestration
// ---------------------------------------------------------------------------
extern "C" void kernel_launch(void* const* d_in, const int* in_sizes, int n_in,
                              void* d_out, int out_size, void* d_ws, size_t ws_size,
                              hipStream_t stream) {
  (void)in_sizes; (void)n_in; (void)out_size; (void)ws_size;

  const float* x     = (const float*)d_in[0];
  const float* wq    = (const float*)d_in[1];
  const float* bq    = (const float*)d_in[2];
  const float* wk    = (const float*)d_in[3];
  const float* bk    = (const float*)d_in[4];
  const float* wv    = (const float*)d_in[5];
  const float* bv    = (const float*)d_in[6];
  const float* wo    = (const float*)d_in[7];
  const float* bo    = (const float*)d_in[8];
  const float* w1    = (const float*)d_in[9];
  const float* b1    = (const float*)d_in[10];
  const float* w2    = (const float*)d_in[11];
  const float* b2    = (const float*)d_in[12];
  const float* g1    = (const float*)d_in[13];
  const float* beta1 = (const float*)d_in[14];
  const float* g2    = (const float*)d_in[15];
  const float* beta2 = (const float*)d_in[16];
  float* out = (float*)d_out;

  char* ws = (char*)d_ws;
  size_t off = 0;
  auto take = [&](size_t bytes) -> void* {
    void* p = ws + off;
    off += (bytes + 255) & ~(size_t)255;
    return p;
  };

  _Float16* xh    = (_Float16*)take((size_t)M_ * D_ * 2);       // x in f16
  _Float16* wqkvT = (_Float16*)take((size_t)3 * D_ * D_ * 2);   // [3D][D]
  _Float16* woT   = (_Float16*)take((size_t)D_ * D_ * 2);       // [D][D]
  _Float16* w1T   = (_Float16*)take((size_t)FF_ * D_ * 2);      // [FF][D]
  _Float16* w2T   = (_Float16*)take((size_t)D_ * FF_ * 2);      // [D][FF]
  _Float16* qh    = (_Float16*)take((size_t)M_ * D_ * 2);       // [B,H,S,HD]
  _Float16* kh    = (_Float16*)take((size_t)M_ * D_ * 2);       // [B,H,S,HD]
  _Float16* vt    = (_Float16*)take((size_t)M_ * D_ * 2);       // [B,H,HD,S]
  float*    attf  = (float*)take((size_t)M_ * D_ * 4);          // o-proj out
  float*    hf32  = (float*)take((size_t)M_ * D_ * 4);          // post-LN1
  _Float16* hf16  = (_Float16*)take((size_t)M_ * D_ * 2);
  _Float16* gh    = (_Float16*)take((size_t)M_ * FF_ * 2);      // gelu(h@w1+b1)
  // Aliases (lifetimes do not overlap):
  _Float16* oh    = xh;    // attention output reuses xh (consumed by QKV GEMM)
  float*    ffnf  = attf;  // FFN2 out reuses attf (consumed by LN1)

  // --- Stage 0: converts / weight transposes ---
  {
    size_t n = (size_t)M_ * D_;
    cvt_f32_f16_kernel<<<(n + 255) / 256, 256, 0, stream>>>(x, xh, n);
    size_t nq = (size_t)3 * D_ * D_;
    qkvw_kernel<<<(nq + 255) / 256, 256, 0, stream>>>(wq, wk, wv, wqkvT);
    size_t nwo = (size_t)D_ * D_;
    transpose_kernel<<<(nwo + 255) / 256, 256, 0, stream>>>(wo, woT, D_, D_);
    size_t nw1 = (size_t)D_ * FF_;
    transpose_kernel<<<(nw1 + 255) / 256, 256, 0, stream>>>(w1, w1T, D_, FF_);
    transpose_kernel<<<(nw1 + 255) / 256, 256, 0, stream>>>(w2, w2T, FF_, D_);
  }

  // --- Stage 1: fused QKV projection ---
  qkv_gemm_kernel<<<dim3(3 * D_ / BN, M_ / BM), 256, 0, stream>>>(
      xh, wqkvT, bq, bk, bv, qh, kh, vt);

  // --- Stage 2: causal flash attention ---
  attn_kernel<<<dim3(S_ / 128, H_, B_), 256, 0, stream>>>(qh, kh, vt, oh);

  // --- Stage 3: output projection ---
  gemm_wmma_kernel<<<dim3(D_ / BN, M_ / BM), 256, 0, stream>>>(
      oh, woT, bo, attf, ((_Float16*)nullptr), M_, D_, D_, 0);

  // --- Stage 4: residual + LN1 ---
  add_ln_kernel<<<M_, 256, 0, stream>>>(x, attf, g1, beta1, hf32, hf16);

  // --- Stage 5: FFN up + GELU ---
  gemm_wmma_kernel<<<dim3(FF_ / BN, M_ / BM), 256, 0, stream>>>(
      hf16, w1T, b1, ((float*)nullptr), gh, M_, FF_, D_, 1);

  // --- Stage 6: FFN down ---
  gemm_wmma_kernel<<<dim3(D_ / BN, M_ / BM), 256, 0, stream>>>(
      gh, w2T, b2, ffnf, ((_Float16*)nullptr), M_, D_, FF_, 0);

  // --- Stage 7: residual + LN2 -> output ---
  add_ln_kernel<<<M_, 256, 0, stream>>>(hf32, ffnf, g2, beta2, out,
                                        ((_Float16*)nullptr));
}
